// MG_GAT_BiLSTM_52218212385020
// MI455X (gfx1250) — matrix operand order repaired and checked by reading, not compiled, not run on hardware
//
#include <hip/hip_runtime.h>
#include <hip/hip_bf16.h>
#include <math.h>

#define Bn 16
#define Tn 96
#define Nn 100
#define EMB 32
#define HEADS 2
#define FPH 16
#define RH 128
#define Mn (Bn*Tn*Nn)       // 153600 tiled nodes
#define COPIES (Bn*Tn)      // 1536 graph copies
#define EPG 2100            // edges per graph copy (2000 + 100 self loops)

typedef __attribute__((ext_vector_type(16))) __bf16 v16bf;
typedef __attribute__((ext_vector_type(8)))  __bf16 v8bf;
typedef __attribute__((ext_vector_type(8)))  float  v8f;
typedef __attribute__((ext_vector_type(4)))  float  v4f;
typedef __attribute__((ext_vector_type(4)))  unsigned v4u;
typedef __attribute__((ext_vector_type(8)))  int    v8i;
typedef __attribute__((ext_vector_type(4)))  int    v4i;

#if defined(__HIP_DEVICE_COMPILE__) && __has_builtin(__builtin_amdgcn_tensor_load_to_lds) && __has_builtin(__builtin_amdgcn_s_wait_tensorcnt)
#define HAVE_TDM 1
#else
#define HAVE_TDM 0
#endif

// ---------- helpers ----------
static __device__ __forceinline__ unsigned short f2bfbits(float f) {
  unsigned u = __float_as_uint(f);
  unsigned r = u + 0x7FFFu + ((u >> 16) & 1u);   // round-to-nearest-even
  return (unsigned short)(r >> 16);
}
static __device__ __forceinline__ float bf2f(unsigned short h) {
  return __uint_as_float((unsigned)h << 16);
}
static __device__ __forceinline__ float rcpf_(float x) { return __builtin_amdgcn_rcpf(x); }
static __device__ __forceinline__ float sigm_(float x) { return rcpf_(1.f + __expf(-x)); }
static __device__ __forceinline__ float tanh_(float x) { return 1.f - 2.f * rcpf_(__expf(2.f * x) + 1.f); }
// order-preserving float <-> uint for atomic segment-max
static __device__ __forceinline__ unsigned f2ord(float f) {
  unsigned u = __float_as_uint(f);
  return (u & 0x80000000u) ? ~u : (u | 0x80000000u);
}
static __device__ __forceinline__ float ord2f(unsigned o) {
  unsigned u = (o & 0x80000000u) ? (o & 0x7FFFFFFFu) : ~o;
  return __uint_as_float(u);
}
static __device__ __forceinline__ v16bf cat16(v8bf lo, v8bf hi) {
  return __builtin_shufflevector(lo, hi, 0,1,2,3,4,5,6,7,8,9,10,11,12,13,14,15);
}

// ---------- node embedding: h0 = x*emb_W^T + emb_b, stored bf16 ----------
__global__ void k_embed(const float* __restrict__ x, const float* __restrict__ eW,
                        const float* __restrict__ eb, unsigned short* __restrict__ xfbf) {
  long i = (long)blockIdx.x * blockDim.x + threadIdx.x;
  if (i >= (long)Mn * EMB) return;
  int m = (int)(i >> 5), c = (int)(i & 31);
  xfbf[i] = f2bfbits(x[m] * eW[c] + eb[c]);
}

// ---------- generic WMMA GEMM:  C[M,Nc] = A_bf16[M,K] @ Wbf16[Nc,K]^T + bias ----------
template<int K>
__global__ void k_gemm_bf16(const unsigned short* __restrict__ A,
                            const unsigned short* __restrict__ Wb,
                            const float* __restrict__ bias,
                            float* __restrict__ C,
                            int Nc, int totTiles) {
  int wv = threadIdx.x >> 5;
  int tile = blockIdx.x * (blockDim.x >> 5) + wv;
  if (tile >= totTiles) return;                  // wave-uniform: EXEC stays all-ones
  int ntN = Nc >> 4;
  int mt = tile / ntN, nt = tile - mt * ntN;
  int lane = threadIdx.x & 31;
  int half = lane >> 4, l15 = lane & 15;
  const __bf16* Ar = (const __bf16*)A + (long)(mt * 16 + l15) * K;
  const __bf16* Wr = (const __bf16*)Wb + (long)(nt * 16 + l15) * K;
  v8f acc = {};
#pragma unroll
  for (int k0 = 0; k0 < K; k0 += 32) {
    __builtin_prefetch(Ar + k0 + 128, 0, 3);     // global_prefetch_b8
    v8bf alo = *(const v8bf*)(Ar + k0 + half * 8);
    v8bf ahi = *(const v8bf*)(Ar + k0 + 16 + half * 8);
    v8bf blo = *(const v8bf*)(Wr + k0 + half * 8);
    v8bf bhi = *(const v8bf*)(Wr + k0 + 16 + half * 8);
    acc = __builtin_amdgcn_wmma_f32_16x16x32_bf16(false, cat16(alo, ahi),
                                                  false, cat16(blo, bhi),
                                                  (short)0, acc, false, false);
  }
#pragma unroll
  for (int r = 0; r < 8; ++r) {
    int m = mt * 16 + r + half * 8;              // C layout: lanes->N, vgpr r -> M=r(+8)
    int n = nt * 16 + l15;
    C[(long)m * Nc + n] = acc[r] + bias[n];
  }
}

// ---------- GAT: init accumulators ----------
__global__ void k_gat_init(float* __restrict__ agg, unsigned* __restrict__ smax,
                           float* __restrict__ denom) {
  long i = (long)blockIdx.x * blockDim.x + threadIdx.x;
  if (i < (long)Mn * EMB) agg[i] = 0.f;
  if (i < (long)Mn * HEADS) { smax[i] = 0x007FFFFFu; denom[i] = 0.f; }   // f2ord(-inf)
}

// ---------- GAT edge pass 1: per-dst segment max (b128 row loads) ----------
__global__ void k_edge_max(const int* __restrict__ ei, const float* __restrict__ xl,
                           const float* __restrict__ xr, const float* __restrict__ att,
                           unsigned* __restrict__ smax) {
  long e = (long)blockIdx.x * blockDim.x + threadIdx.x;
  if (e >= (long)EPG * COPIES) return;
  int j = (int)(e % EPG), g = (int)(e / EPG);
  int src = ei[j] + g * Nn;
  int dst = ei[EPG + j] + g * Nn;
  const v4f* a4 = (const v4f*)(xl + (long)src * EMB);
  const v4f* b4 = (const v4f*)(xr + (long)dst * EMB);
#pragma unroll
  for (int h = 0; h < HEADS; ++h) {
    float sc = 0.f;
#pragma unroll
    for (int q = 0; q < 4; ++q) {
      v4f v = a4[h * 4 + q] + b4[h * 4 + q];
#pragma unroll
      for (int c = 0; c < 4; ++c) {
        float u = v[c];
        u = u > 0.f ? u : 0.2f * u;              // leaky_relu(0.2)
        sc += u * att[h * FPH + q * 4 + c];
      }
    }
    atomicMax(&smax[(long)dst * HEADS + h], f2ord(sc));
  }
}

// ---------- GAT edge pass 2: denom += e, agg += e * xl[src] ----------
__global__ void k_edge_acc(const int* __restrict__ ei, const float* __restrict__ xl,
                           const float* __restrict__ xr, const float* __restrict__ att,
                           const unsigned* __restrict__ smax,
                           float* __restrict__ denom, float* __restrict__ agg) {
  long e = (long)blockIdx.x * blockDim.x + threadIdx.x;
  if (e >= (long)EPG * COPIES) return;
  int j = (int)(e % EPG), g = (int)(e / EPG);
  int src = ei[j] + g * Nn;
  int dst = ei[EPG + j] + g * Nn;
  const v4f* a4 = (const v4f*)(xl + (long)src * EMB);
  const v4f* b4 = (const v4f*)(xr + (long)dst * EMB);
  v4f av[8];
#pragma unroll
  for (int q = 0; q < 8; ++q) av[q] = a4[q];
#pragma unroll
  for (int h = 0; h < HEADS; ++h) {
    float sc = 0.f;
#pragma unroll
    for (int q = 0; q < 4; ++q) {
      v4f v = av[h * 4 + q] + b4[h * 4 + q];
#pragma unroll
      for (int c = 0; c < 4; ++c) {
        float u = v[c];
        u = u > 0.f ? u : 0.2f * u;
        sc += u * att[h * FPH + q * 4 + c];
      }
    }
    float a = __expf(sc - ord2f(smax[(long)dst * HEADS + h]));
    atomicAdd(&denom[(long)dst * HEADS + h], a);
#pragma unroll
    for (int q = 0; q < 4; ++q)
#pragma unroll
      for (int c = 0; c < 4; ++c)
        atomicAdd(&agg[(long)dst * EMB + h * FPH + q * 4 + c], av[h * 4 + q][c] * a);
  }
}

// ---------- GAT finalize: h = elu(agg/denom + bias), f32 + bf16 ----------
__global__ void k_gat_fin(const float* __restrict__ agg, const float* __restrict__ denom,
                          const float* __restrict__ bias,
                          float* __restrict__ outf, unsigned short* __restrict__ outbf) {
  long i = (long)blockIdx.x * blockDim.x + threadIdx.x;
  if (i >= (long)Mn * EMB) return;
  int m = (int)(i >> 5), c = (int)(i & 31), h = c >> 4;
  float v = agg[i] * rcpf_(denom[(long)m * HEADS + h]) + bias[c];
  v = v > 0.f ? v : (__expf(v) - 1.f);            // elu
  outf[i] = v;
  outbf[i] = f2bfbits(v);
}

// ---------- branch gating: hsum = (br==0 ? 0 : hsum) + softmax(gate)[br]*h ----------
__global__ void k_accum(const float* __restrict__ hb, const float* __restrict__ gate,
                        float* __restrict__ hsum, int brIdx) {
  long i = (long)blockIdx.x * blockDim.x + threadIdx.x;
  if (i >= (long)Mn * EMB) return;
  float g0 = gate[0], g1 = gate[1], g2 = gate[2];
  float mx = fmaxf(g0, fmaxf(g1, g2));
  float e0 = __expf(g0 - mx), e1 = __expf(g1 - mx), e2 = __expf(g2 - mx);
  float w = (brIdx == 0 ? e0 : brIdx == 1 ? e1 : e2) * rcpf_(e0 + e1 + e2);
  float v = w * hb[i];
  hsum[i] = (brIdx == 0) ? v : (hsum[i] + v);
}

// ---------- channel attention ----------
__global__ void k_camean(const float* __restrict__ hsum, float* __restrict__ sbuf) {
  int i = blockIdx.x * blockDim.x + threadIdx.x;
  if (i >= COPIES * EMB) return;
  int bt = i >> 5, c = i & 31;
  float s = 0.f;
  for (int n = 0; n < Nn; ++n) s += hsum[((long)bt * Nn + n) * EMB + c];
  sbuf[i] = s * (1.f / Nn);
}
__global__ void k_camlp(const float* __restrict__ sbuf, const float* __restrict__ W1,
                        const float* __restrict__ W2, float* __restrict__ cw) {
  int bt = blockIdx.x * blockDim.x + threadIdx.x;
  if (bt >= COPIES) return;
  float hid[8];
#pragma unroll
  for (int k = 0; k < 8; ++k) {
    float a = 0.f;
    for (int j = 0; j < 32; ++j) a += sbuf[bt * 32 + j] * W1[k * 32 + j];
    hid[k] = fmaxf(a, 0.f);
  }
  for (int c = 0; c < 32; ++c) {
    float a = 0.f;
#pragma unroll
    for (int k = 0; k < 8; ++k) a += hid[k] * W2[c * 8 + k];
    cw[bt * 32 + c] = sigm_(a);
  }
}
__global__ void k_applycw(float* __restrict__ hsum, const float* __restrict__ cw) {
  long i = (long)blockIdx.x * blockDim.x + threadIdx.x;
  if (i >= (long)Mn * EMB) return;
  int m = (int)(i >> 5), c = (int)(i & 31);
  hsum[i] *= cw[(m / Nn) * 32 + c];
}

// ---------- spatial attention: softmax over nodes of h@sa_w ----------
__global__ void k_spatial(const float* __restrict__ hsum, const float* __restrict__ saw,
                          float* __restrict__ sw) {
  int bt = blockIdx.x * blockDim.x + threadIdx.x;
  if (bt >= COPIES) return;
  float lg[Nn]; float mx = -1e30f;
  for (int n = 0; n < Nn; ++n) {
    const v4f* p = (const v4f*)(hsum + ((long)bt * Nn + n) * EMB);
    float a = 0.f;
#pragma unroll
    for (int q = 0; q < 8; ++q) {
      v4f v = p[q];
#pragma unroll
      for (int c = 0; c < 4; ++c) a += v[c] * saw[q * 4 + c];
    }
    lg[n] = a; mx = fmaxf(mx, a);
  }
  float s = 0.f;
  for (int n = 0; n < Nn; ++n) { lg[n] = __expf(lg[n] - mx); s += lg[n]; }
  float inv = rcpf_(s);
  for (int n = 0; n < Nn; ++n) sw[(long)bt * Nn + n] = lg[n] * inv;
}

// ---------- apply sw + transpose [B,T,N,E] -> seq_bf16 [T,B,N*E] ----------
__global__ void k_pack(const float* __restrict__ hsum, const float* __restrict__ sw,
                       unsigned short* __restrict__ seqbf) {
  long i = (long)blockIdx.x * blockDim.x + threadIdx.x;
  if (i >= (long)Mn * EMB) return;
  int m = (int)(i >> 5), c = (int)(i & 31);
  int bt = m / Nn, n = m - bt * Nn;
  int b = bt / Tn, t = bt - b * Tn;
  float v = hsum[i] * sw[(long)bt * Nn + n];
  seqbf[((long)t * Bn + b) * (Nn * EMB) + n * EMB + c] = f2bfbits(v);
}

// ---------- small utilities ----------
__global__ void k_addvec(const float* a, const float* b, float* o, int n) {
  int i = blockIdx.x * blockDim.x + threadIdx.x;
  if (i < n) o[i] = a[i] + b[i];
}
__global__ void k_cvtbf(const float* a, unsigned short* o, int n) {
  int i = blockIdx.x * blockDim.x + threadIdx.x;
  if (i < n) o[i] = f2bfbits(a[i]);
}

// ---------- LSTM recurrence: one block/direction, WMMA on LDS h, TDM-staged zin ----------
__global__ void k_lstm_rec(const float* __restrict__ zin0, const float* __restrict__ zin1,
                           const unsigned short* __restrict__ whh0,
                           const unsigned short* __restrict__ whh1,
                           float* __restrict__ outf, unsigned short* __restrict__ outbf) {
  __shared__ unsigned short hls[Bn * RH];          // h state, bf16 (4KB)
  __shared__ unsigned short zpre[Bn * 4 * RH];     // recurrent pre-activations, bf16 (16KB)
  __shared__ float zbuf[Bn * 4 * RH];              // TDM-staged zin[t] slice, f32 (32KB)
  int dir = blockIdx.x;
  const float* zin = dir ? zin1 : zin0;
  const __bf16* Whh = (const __bf16*)(dir ? whh1 : whh0);
  for (int i = threadIdx.x; i < Bn * RH; i += blockDim.x) hls[i] = 0;
  float creg[4] = {0.f, 0.f, 0.f, 0.f};            // c state in registers (4 cells/thread)
  __syncthreads();
  int lane = threadIdx.x & 31, wv = threadIdx.x >> 5;
  int half = lane >> 4, l15 = lane & 15;
  const __bf16* Ab = (const __bf16*)hls + l15 * RH;
  for (int s = 0; s < Tn; ++s) {
    int t = dir ? (Tn - 1 - s) : s;
#if HAVE_TDM
    // Tensor Data Mover: stage zin[t] (16x512 f32 = 32KB, contiguous 1-D tile) into LDS
    // while the recurrent WMMAs execute. D#: count=1, data_size=4B, dims/tile = 8192.
    if (wv == 0) {
      unsigned long long ga = (unsigned long long)(const void*)(zin + (long)t * Bn * 512);
      unsigned lds = (unsigned)(unsigned long long)(const void*)zbuf;
      v4u g0 = { 1u, lds, (unsigned)ga,
                 (unsigned)((ga >> 32) & 0x1FFFFFFu) | 0x80000000u };  // type=2 (image)
      v8i g1 = { 0x20000,                 // data_size=2 (4B), no mask/pad/iterate
                 0x2000 << 16,            // tensor_dim0 = 8192 (low 16)
                 1 << 16,                 // tensor_dim0 hi=0, tensor_dim1 = 1
                 0x2000 << 16,            // tile_dim0 = 8192
                 1,                       // tile_dim1 = 1
                 0x2000, 0, 0 };          // tensor_dim0_stride = 8192
      v4i gz4 = { 0, 0, 0, 0 };
      v8i gz8 = { 0, 0, 0, 0, 0, 0, 0, 0 };
      __builtin_amdgcn_tensor_load_to_lds(g0, g1, gz4, gz4, gz8, 0);
    }
#endif
    // zr = h @ Whh^T : M=16(batch), K=128, N=512 -> 16 waves x 2 tiles x 4 K-steps
#pragma unroll
    for (int q = 0; q < 2; ++q) {
      int nt = wv * 2 + q;
      int bcol = nt * 16 + l15;
      const __bf16* Wr = Whh + (long)bcol * RH;
      v8f acc = {};
#pragma unroll
      for (int k0 = 0; k0 < RH; k0 += 32) {
        v8bf alo = *(const v8bf*)(Ab + k0 + half * 8);       // ds_load_b128
        v8bf ahi = *(const v8bf*)(Ab + k0 + 16 + half * 8);
        v8bf blo = *(const v8bf*)(Wr + k0 + half * 8);
        v8bf bhi = *(const v8bf*)(Wr + k0 + 16 + half * 8);
        acc = __builtin_amdgcn_wmma_f32_16x16x32_bf16(false, cat16(alo, ahi),
                                                      false, cat16(blo, bhi),
                                                      (short)0, acc, false, false);
      }
#pragma unroll
      for (int r = 0; r < 8; ++r) zpre[(r + half * 8) * 512 + bcol] = f2bfbits(acc[r]);
    }
#if HAVE_TDM
    if (wv == 0) __builtin_amdgcn_s_wait_tensorcnt(0);
    __syncthreads();
    const float* zt = zbuf;
#else
    __syncthreads();
    const float* zt = zin + (long)t * Bn * 512;
#endif
#pragma unroll
    for (int kk = 0; kk < 4; ++kk) {
      int idx = kk * 512 + threadIdx.x;              // blockDim == 512
      int b = idx >> 7, j = idx & 127;
      const float* zb = zt + b * 512;
      const unsigned short* zp = zpre + b * 512;
      float iv = sigm_(zb[j]          + bf2f(zp[j]));
      float fv = sigm_(zb[RH + j]     + bf2f(zp[RH + j]));
      float gv = tanh_(zb[2 * RH + j] + bf2f(zp[2 * RH + j]));
      float ov = sigm_(zb[3 * RH + j] + bf2f(zp[3 * RH + j]));
      float cc = fv * creg[kk] + iv * gv;
      float hh = ov * tanh_(cc);
      creg[kk] = cc;
      hls[idx] = f2bfbits(hh);
      long off = ((long)t * Bn + b) * (2 * RH) + (long)dir * RH + j;
      outf[off] = hh;
      outbf[off] = f2bfbits(hh);
    }
    __syncthreads();
  }
}

// ---------- head: out[b,n] = seq3[T-1,b,:] @ head_W[n,:] + head_b[n] ----------
__global__ void k_head(const float* __restrict__ seq3, const float* __restrict__ hW,
                       const float* __restrict__ hb, float* __restrict__ out) {
  int i = blockIdx.x * blockDim.x + threadIdx.x;
  if (i >= Bn * Nn) return;
  int b = i / Nn, n = i - b * Nn;
  const v4f* p = (const v4f*)(seq3 + ((long)(Tn - 1) * Bn + b) * (2 * RH));
  const v4f* w4 = (const v4f*)(hW + (long)n * 2 * RH);
  float a = hb[n];
  for (int q = 0; q < (2 * RH) / 4; ++q) {
    v4f pv = p[q], wvv = w4[q];
#pragma unroll
    for (int c = 0; c < 4; ++c) a += pv[c] * wvv[c];
  }
  out[b * Nn + n] = a;
}

extern "C" void kernel_launch(void* const* d_in, const int* in_sizes, int n_in,
                              void* d_out, int out_size, void* d_ws, size_t ws_size,
                              hipStream_t stream) {
  (void)in_sizes; (void)n_in; (void)out_size; (void)ws_size;
  const float* x = (const float*)d_in[0];
  const int* ei[3] = { (const int*)d_in[1], (const int*)d_in[3], (const int*)d_in[5] };
  auto pf = [&](int i) { return (const float*)d_in[i]; };
  // params flattened in dict order starting at index 7
  const float* embW = pf(7);  const float* embB = pf(8);
  const float* gate = pf(45);
  const float* caW1 = pf(46); const float* caW2 = pf(47); const float* saw = pf(48);
  const float* headW = pf(65); const float* headB = pf(66);

  // workspace carve-out
  char* ws = (char*)d_ws;
  size_t off = 0;
  auto alloc = [&](size_t bytes) -> char* {
    char* p = ws + off; off += (bytes + 255) & ~(size_t)255; return p;
  };
  unsigned short* xfbf  = (unsigned short*)alloc((size_t)Mn * EMB * 2);
  unsigned short* hbf   = (unsigned short*)alloc((size_t)Mn * EMB * 2);
  float* xl    = (float*)alloc((size_t)Mn * EMB * 4);
  float* xr    = (float*)alloc((size_t)Mn * EMB * 4);
  float* agg   = (float*)alloc((size_t)Mn * EMB * 4);
  float* hsum  = (float*)alloc((size_t)Mn * EMB * 4);
  unsigned* smax = (unsigned*)alloc((size_t)Mn * HEADS * 4);
  float* denom = (float*)alloc((size_t)Mn * HEADS * 4);
  float* sbuf  = (float*)alloc((size_t)COPIES * EMB * 4);
  float* cw    = (float*)alloc((size_t)COPIES * EMB * 4);
  float* sw    = (float*)alloc((size_t)COPIES * Nn * 4);
  unsigned short* seqbf = (unsigned short*)alloc((size_t)COPIES * Nn * EMB * 2);
  float* zin0  = (float*)alloc((size_t)COPIES * 512 * 4);
  float* zin1  = (float*)alloc((size_t)COPIES * 512 * 4);
  float* seq2f = (float*)alloc((size_t)COPIES * 256 * 4);
  unsigned short* seq2bf = (unsigned short*)alloc((size_t)COPIES * 256 * 2);
  float* seq3f = (float*)alloc((size_t)COPIES * 256 * 4);
  unsigned short* seq3bf = (unsigned short*)alloc((size_t)COPIES * 256 * 2);
  // bf16 weight copies (one-time conversion, L2-resident)
  unsigned short* wlrbf = (unsigned short*)alloc((size_t)12 * EMB * EMB * 2);   // 6 convs x (Wl,Wr)
  unsigned short* whhbf = (unsigned short*)alloc((size_t)4 * 512 * RH * 2);
  unsigned short* wih0bf = (unsigned short*)alloc((size_t)2 * 512 * Nn * EMB * 2);
  unsigned short* wih1bf = (unsigned short*)alloc((size_t)2 * 512 * 2 * RH * 2);
  float* cb    = (float*)alloc((size_t)4 * 512 * 4);

  const int TPB = 256;
  const long nElem = (long)Mn * EMB;                       // 4,915,200
  const int gElem = (int)((nElem + TPB - 1) / TPB);
  const long nE = (long)EPG * COPIES;                      // 3,225,600 edges
  const int gE = (int)((nE + TPB - 1) / TPB);

  // ---- one-time weight conversions to bf16 ----
  for (int cv6 = 0; cv6 < 6; ++cv6) {
    int base = 9 + cv6 * 6;
    k_cvtbf<<<(EMB * EMB + TPB - 1) / TPB, TPB, 0, stream>>>(pf(base + 0), wlrbf + (size_t)(2 * cv6 + 0) * EMB * EMB, EMB * EMB);
    k_cvtbf<<<(EMB * EMB + TPB - 1) / TPB, TPB, 0, stream>>>(pf(base + 2), wlrbf + (size_t)(2 * cv6 + 1) * EMB * EMB, EMB * EMB);
  }
  for (int ld = 0; ld < 4; ++ld) {
    int base = 49 + ld * 4;                                // Wih, Whh, bih, bhh
    k_addvec<<<2, TPB, 0, stream>>>(pf(base + 2), pf(base + 3), cb + ld * 512, 512);
    k_cvtbf<<<(512 * RH + TPB - 1) / TPB, TPB, 0, stream>>>(pf(base + 1), whhbf + (size_t)ld * 512 * RH, 512 * RH);
  }
  {
    int n1 = 512 * Nn * EMB;                               // 1,638,400
    k_cvtbf<<<(n1 + TPB - 1) / TPB, TPB, 0, stream>>>(pf(49), wih0bf, n1);
    k_cvtbf<<<(n1 + TPB - 1) / TPB, TPB, 0, stream>>>(pf(53), wih0bf + (size_t)n1, n1);
    int n2 = 512 * 2 * RH;                                 // 131,072
    k_cvtbf<<<(n2 + TPB - 1) / TPB, TPB, 0, stream>>>(pf(57), wih1bf, n2);
    k_cvtbf<<<(n2 + TPB - 1) / TPB, TPB, 0, stream>>>(pf(61), wih1bf + (size_t)n2, n2);
  }

  k_embed<<<gElem, TPB, 0, stream>>>(x, embW, embB, xfbf);

  // ---- three GATv2 branches, two convs each ----
  for (int br = 0; br < 3; ++br) {
    const unsigned short* cur = xfbf;
    for (int cv = 0; cv < 2; ++cv) {
      int cv6 = br * 2 + cv;
      int base = 9 + cv6 * 6;
      const float* bl   = pf(base + 1);
      const float* brr  = pf(base + 3);
      const float* att  = pf(base + 4);
      const float* bias = pf(base + 5);
      const unsigned short* Wlb = wlrbf + (size_t)(2 * cv6 + 0) * EMB * EMB;
      const unsigned short* Wrb = wlrbf + (size_t)(2 * cv6 + 1) * EMB * EMB;
      int tiles = (Mn / 16) * (EMB / 16);                  // 19200
      int gg = (tiles + 3) / 4;
      k_gemm_bf16<EMB><<<gg, 128, 0, stream>>>(cur, Wlb, bl, xl, EMB, tiles);
      k_gemm_bf16<EMB><<<gg, 128, 0, stream>>>(cur, Wrb, brr, xr, EMB, tiles);
      k_gat_init<<<gElem, TPB, 0, stream>>>(agg, smax, denom);
      k_edge_max<<<gE, TPB, 0, stream>>>(ei[br], xl, xr, att, smax);
      k_edge_acc<<<gE, TPB, 0, stream>>>(ei[br], xl, xr, att, smax, denom, agg);
      k_gat_fin<<<gElem, TPB, 0, stream>>>(agg, denom, bias, xl /*f32 h*/, hbf);
      cur = hbf;
    }
    k_accum<<<gElem, TPB, 0, stream>>>(xl, gate, hsum, br);
  }

  // ---- channel + spatial attention, pack into [T,B,3200] bf16 ----
  k_camean<<<(COPIES * EMB + TPB - 1) / TPB, TPB, 0, stream>>>(hsum, sbuf);
  k_camlp<<<(COPIES + TPB - 1) / TPB, TPB, 0, stream>>>(sbuf, caW1, caW2, cw);
  k_applycw<<<gElem, TPB, 0, stream>>>(hsum, cw);
  k_spatial<<<(COPIES + TPB - 1) / TPB, TPB, 0, stream>>>(hsum, saw, sw);
  k_pack<<<gElem, TPB, 0, stream>>>(hsum, sw, seqbf);

  // ---- BiLSTM layer 1: time-parallel input projections (K=3200) then recurrence ----
  {
    int tiles = (COPIES / 16) * (512 / 16);                // 3072
    int gg = (tiles + 3) / 4;
    k_gemm_bf16<Nn * EMB><<<gg, 128, 0, stream>>>(seqbf, wih0bf, cb + 0 * 512, zin0, 512, tiles);
    k_gemm_bf16<Nn * EMB><<<gg, 128, 0, stream>>>(seqbf, wih0bf + (size_t)512 * Nn * EMB, cb + 1 * 512, zin1, 512, tiles);
    k_lstm_rec<<<2, 512, 0, stream>>>(zin0, zin1, whhbf, whhbf + 512 * RH, seq2f, seq2bf);
  }
  // ---- BiLSTM layer 2 (K=256) ----
  {
    int tiles = (COPIES / 16) * (512 / 16);
    int gg = (tiles + 3) / 4;
    k_gemm_bf16<2 * RH><<<gg, 128, 0, stream>>>(seq2bf, wih1bf, cb + 2 * 512, zin0, 512, tiles);
    k_gemm_bf16<2 * RH><<<gg, 128, 0, stream>>>(seq2bf, wih1bf + (size_t)512 * 2 * RH, cb + 3 * 512, zin1, 512, tiles);
    k_lstm_rec<<<2, 512, 0, stream>>>(zin0, zin1, whhbf + (size_t)2 * 512 * RH, whhbf + (size_t)3 * 512 * RH, seq3f, seq3bf);
  }

  k_head<<<(Bn * Nn + TPB - 1) / TPB, TPB, 0, stream>>>(seq3f, headW, headB, (float*)d_out);
}
